// WESUP_82910048682002
// MI455X (gfx1250) — matrix operand order repaired and checked by reading, not compiled, not run on hardware
//
#include <hip/hip_runtime.h>
#include <hip/hip_bf16.h>
#include <math.h>

// ---------------------------------------------------------------------------
// WESUP pipeline for MI455X (gfx1250, wave32, WMMA).
// All heavy matmuls run through one bf16-WMMA tiled GEMM (v_wmma_f32_16x16x32_bf16).
// LDS tiles stored as packed bf16-pairs (uint32): fragments = aligned ds_load_b128.
// Global tile loads are software-pipelined one iteration ahead of the WMMAs.
// ---------------------------------------------------------------------------

typedef __attribute__((ext_vector_type(16))) __bf16 v16bf;
typedef __attribute__((ext_vector_type(8)))  float  v8f;

#define HW_   50176   // 224*224
#define CSUM_ 2112
#define NSP_  1000
#define NL_   200
#define NB_   100
#define TRN_  400
#define INT_  1024
#define DCLS_ 32
#define DREL_ 512

// ---- f32 -> packed bf16 pair (RNE) ----------------------------------------
#if defined(__has_builtin) && __has_builtin(__builtin_amdgcn_cvt_pk_bf16_f32)
__device__ __forceinline__ unsigned int packbf(float lo, float hi) {
  auto r = __builtin_amdgcn_cvt_pk_bf16_f32(lo, hi);   // v_cvt_pk_bf16_f32
  union { decltype(r) v; unsigned int u; } cv; cv.v = r;
  return cv.u;
}
#else
__device__ __forceinline__ unsigned int bfbits(float f) {
  union { float f; unsigned int u; } v; v.f = f;
  return (v.u + 0x7FFFu + ((v.u >> 16) & 1u)) >> 16;
}
__device__ __forceinline__ unsigned int packbf(float lo, float hi) {
  return bfbits(lo) | (bfbits(hi) << 16);
}
#endif

// ---------------------------------------------------------------------------
// Tiled GEMM: C[M,N] = relu(A[M,K] @ W[K,N] + bias[N]),  K % 32 == 0, N % 4 == 0.
// Block tile 128x64, BK=32, 256 threads (8 waves, 4x2), 2x2 WMMA frags/wave.
// ---------------------------------------------------------------------------
#define BM 128
#define BN 64
#define BK 32
#define LDAU 20   // uint32 row stride, A tile [row][kpair], padded to 16B multiple
#define LDBU 20   // uint32 row stride, B tile [col][kpair]

union FragAB { v16bf v; unsigned int u32[8]; };

__global__ __launch_bounds__(256)
void gemm_bias_relu(const float* __restrict__ A, const float* __restrict__ W,
                    const float* __restrict__ bias, float* __restrict__ C,
                    int M, int N, int K) {
  __shared__ unsigned int As[BM * LDAU];
  __shared__ unsigned int Bs[BN * LDBU];

  const int tid  = threadIdx.x;
  const int lane = tid & 31;
  const int wave = tid >> 5;
  const int wm   = wave >> 1;        // 0..3  (rows, 32 each)
  const int wn   = wave & 1;         // 0..1  (cols, 32 each)

  const int row0 = blockIdx.y * BM;
  const int col0 = blockIdx.x * BN;

  const int mi = lane & 15;          // frag row/col within 16
  const int kg = lane >> 4;          // K-half select

  v8f acc00 = {}, acc01 = {}, acc10 = {}, acc11 = {};

  // cooperative-staging indices
  const int arow  = tid >> 1;                   // 0..127
  const int apair = (tid & 1) * 8;              // kpair offset 0 / 8 (k 0 / 16)
  const int agr   = min(row0 + arow, M - 1);    // clamped: OOB rows only feed
                                                // store-guarded outputs
  const int bn    = tid & 63;                   // 0..63 tile column
  const int kseg  = (tid >> 6) * 4;             // kpairs kseg..kseg+3
  const int bgc   = min(col0 + bn, N - 1);      // clamped column

  const float* aBase = A + (size_t)agr * K + apair * 2;
  const float* wBase = W + (size_t)(2 * kseg) * N + bgc;

  // ---- pipeline prologue: load tile kt=0 into registers ----
  float4 la[4];
  float  lb[8];
  {
    const float4* ap4 = (const float4*)aBase;
    #pragma unroll
    for (int j = 0; j < 4; ++j) la[j] = ap4[j];
    #pragma unroll
    for (int j = 0; j < 4; ++j) {
      lb[2 * j]     = wBase[(size_t)(2 * j) * N];
      lb[2 * j + 1] = wBase[(size_t)(2 * j + 1) * N];
    }
  }

  for (int kt = 0; kt < K; kt += BK) {
    // ---- pack staged registers into LDS (v_cvt_pk_bf16_f32 when available) ----
    {
      uint4* dst = (uint4*)&As[arow * LDAU + apair];
      dst[0] = make_uint4(packbf(la[0].x, la[0].y), packbf(la[0].z, la[0].w),
                          packbf(la[1].x, la[1].y), packbf(la[1].z, la[1].w));
      dst[1] = make_uint4(packbf(la[2].x, la[2].y), packbf(la[2].z, la[2].w),
                          packbf(la[3].x, la[3].y), packbf(la[3].z, la[3].w));
      *(uint4*)&Bs[bn * LDBU + kseg] =
          make_uint4(packbf(lb[0], lb[1]), packbf(lb[2], lb[3]),
                     packbf(lb[4], lb[5]), packbf(lb[6], lb[7]));
    }
    __syncthreads();

    // ---- issue next tile's global loads (overlap with WMMAs below) ----
    {
      const int ktn = (kt + BK < K) ? (kt + BK) : kt;  // branchless tail reload
      const float4* ap4 = (const float4*)(aBase + ktn);
      #pragma unroll
      for (int j = 0; j < 4; ++j) la[j] = ap4[j];
      const float* wp = wBase + (size_t)ktn * N;
      #pragma unroll
      for (int j = 0; j < 4; ++j) {
        lb[2 * j]     = wp[(size_t)(2 * j) * N];
        lb[2 * j + 1] = wp[(size_t)(2 * j + 1) * N];
      }
      // warm L2 two tiles ahead (global_prefetch_b8)
      __builtin_prefetch(aBase + min(kt + 2 * BK, K - BK), 0, 1);
    }

    // ---- fragments: two aligned uint4 LDS reads each ----
    FragAB a0, a1, b0, b1;
    {
      const unsigned int* r0 = &As[(wm * 32 + mi) * LDAU];
      const unsigned int* r1 = &As[(wm * 32 + 16 + mi) * LDAU];
      *(uint4*)&a0.u32[0] = *(const uint4*)(r0 + kg * 4);
      *(uint4*)&a0.u32[4] = *(const uint4*)(r0 + 8 + kg * 4);
      *(uint4*)&a1.u32[0] = *(const uint4*)(r1 + kg * 4);
      *(uint4*)&a1.u32[4] = *(const uint4*)(r1 + 8 + kg * 4);
      const unsigned int* c0 = &Bs[(wn * 32 + mi) * LDBU + kg * 4];
      const unsigned int* c1 = &Bs[(wn * 32 + 16 + mi) * LDBU + kg * 4];
      *(uint4*)&b0.u32[0] = *(const uint4*)(c0);
      *(uint4*)&b0.u32[4] = *(const uint4*)(c0 + 8);
      *(uint4*)&b1.u32[0] = *(const uint4*)(c1);
      *(uint4*)&b1.u32[4] = *(const uint4*)(c1 + 8);
    }

    acc00 = __builtin_amdgcn_wmma_f32_16x16x32_bf16(false, a0.v, false, b0.v,
                                                    (short)0, acc00, false, false);
    acc01 = __builtin_amdgcn_wmma_f32_16x16x32_bf16(false, a0.v, false, b1.v,
                                                    (short)0, acc01, false, false);
    acc10 = __builtin_amdgcn_wmma_f32_16x16x32_bf16(false, a1.v, false, b0.v,
                                                    (short)0, acc10, false, false);
    acc11 = __builtin_amdgcn_wmma_f32_16x16x32_bf16(false, a1.v, false, b1.v,
                                                    (short)0, acc11, false, false);
    __syncthreads();
  }

  // ---- epilogue: bias + relu ----
  // C/D VGPR layout: n = lane&15, rows v + 8*(lane>=16) per 16x16 tile.
  const int half = lane >> 4;
  const int nl   = lane & 15;
  const bool interior = (row0 + BM <= M) & (col0 + BN <= N);

  #pragma unroll
  for (int ti = 0; ti < 2; ++ti) {
    #pragma unroll
    for (int tj = 0; tj < 2; ++tj) {
      v8f acc = (ti == 0) ? (tj == 0 ? acc00 : acc01)
                          : (tj == 0 ? acc10 : acc11);
      const int col   = col0 + wn * 32 + tj * 16 + nl;
      const int rbase = row0 + wm * 32 + ti * 16 + half * 8;
      if (interior) {
        // fast path: base pointer once; v*N folds into the 24-bit inst offset
        const float b = bias[col];
        float* cp = C + (size_t)rbase * N + col;
        #pragma unroll
        for (int v = 0; v < 8; ++v) {
          float x = acc[v] + b;
          cp[(size_t)v * N] = x > 0.0f ? x : 0.0f;
        }
      } else if (col < N) {
        const float b = bias[col];
        float* cp = C + (size_t)rbase * N + col;
        #pragma unroll
        for (int v = 0; v < 8; ++v) {
          if (rbase + v < M) {
            float x = acc[v] + b;
            cp[(size_t)v * N] = x > 0.0f ? x : 0.0f;
          }
        }
      }
    }
  }
}

// ---------------------------------------------------------------------------
// 2-class head: out[i] = softmax(h[i] @ w2 + b2)[:,1]
// ---------------------------------------------------------------------------
__global__ void head2_softmax(const float* __restrict__ H, const float* __restrict__ w2,
                              const float* __restrict__ b2, float* __restrict__ out,
                              int M, int D) {
  int i = blockIdx.x * blockDim.x + threadIdx.x;
  if (i >= M) return;
  const float* h = H + (size_t)i * D;
  float l0 = b2[0], l1 = b2[1];
  for (int c = 0; c < D; ++c) {
    float x = h[c];
    l0 = fmaf(x, w2[c * 2 + 0], l0);
    l1 = fmaf(x, w2[c * 2 + 1], l1);
  }
  out[i] = 1.0f / (1.0f + __expf(l0 - l1));
}

__global__ void zero_f32(float* __restrict__ p, int n) {
  int i = blockIdx.x * blockDim.x + threadIdx.x;
  if (i < n) p[i] = 0.0f;
}

// wave-per-pixel segment accumulation: lane covers 16 channels of 512
__global__ __launch_bounds__(256)
void seg_accum(const float* __restrict__ RF, const int* __restrict__ sp,
               float* __restrict__ SP, float* __restrict__ CNT) {
  int p = blockIdx.x * 8 + (threadIdx.x >> 5);
  int lane = threadIdx.x & 31;
  if (p >= HW_) return;
  int s = sp[p];
  const float* rf = RF + (size_t)p * DREL_;
  float* dst = SP + (size_t)s * DREL_;
  #pragma unroll
  for (int j = 0; j < 16; ++j) {
    int c = lane * 16 + j;
    atomicAdd(&dst[c], rf[c]);
  }
  if (lane == 0) atomicAdd(&CNT[s], 1.0f);
}

__global__ void seg_div(float* __restrict__ SP, const float* __restrict__ CNT, int n) {
  int i = blockIdx.x * blockDim.x + threadIdx.x;
  if (i >= n) return;
  float c = CNT[i / DREL_];
  SP[i] /= (c > 1.0f ? c : 1.0f);
}

__global__ void gather_base(const float* __restrict__ SP, const int* __restrict__ bidx,
                            float* __restrict__ BF) {
  int i = blockIdx.x * blockDim.x + threadIdx.x;   // over 100*512
  if (i >= NB_ * DREL_) return;
  int r = i / DREL_, c = i % DREL_;
  BF[i] = SP[(size_t)bidx[r] * DREL_ + c];
}

// CC[i] = concat(BF[sel[i]], SP[sec[i]])  (400 x 1024)
__global__ void concat_rel(const float* __restrict__ BF, const float* __restrict__ SP,
                           const int* __restrict__ sel, const int* __restrict__ sec,
                           float* __restrict__ CC) {
  int i = blockIdx.x * blockDim.x + threadIdx.x;   // over 400*1024
  if (i >= TRN_ * 2 * DREL_) return;
  int r = i >> 10, c = i & 1023;
  CC[i] = (c < DREL_) ? BF[(size_t)sel[r] * DREL_ + c]
                      : SP[(size_t)sec[r] * DREL_ + (c - DREL_)];
}

// CC[i] = concat(BF[idx[i]], RF[i])  (HW x 1024), float4-vectorized
__global__ void concat_pix(const float4* __restrict__ BF, const float4* __restrict__ RF,
                           const int* __restrict__ idx, float4* __restrict__ CC) {
  long long i = (long long)blockIdx.x * blockDim.x + threadIdx.x;  // over HW*256 float4
  if (i >= (long long)HW_ * 256) return;
  int r = (int)(i >> 8), c = (int)(i & 255);   // c in float4 units, 256/row
  CC[i] = (c < 128) ? BF[(size_t)idx[r] * 128 + c]
                    : RF[(size_t)r * 128 + (c - 128)];
}

__global__ void rel_target(const float* __restrict__ sp_labels,
                           const int* __restrict__ sec, float* __restrict__ out) {
  int i = blockIdx.x * blockDim.x + threadIdx.x;
  if (i >= TRN_) return;
  out[i] = sp_labels[sec[i] * 2 + 1];
}

__global__ void pseudo_labels(const int* __restrict__ sp, const float* __restrict__ sp_labels,
                              const float* __restrict__ P1, const float* __restrict__ P2,
                              float* __restrict__ out) {
  int i = blockIdx.x * blockDim.x + threadIdx.x;
  if (i >= HW_) return;
  int s = sp[i];
  bool labeled = s < NL_;
  int sc = s < 0 ? 0 : (s > NL_ - 1 ? NL_ - 1 : s);
  float ps = labeled ? sp_labels[sc * 2 + 1] : 0.0f;
  bool unl = !labeled;
  float p1 = P1[i];
  bool pos1 = p1 > 0.8f, neg1 = p1 < 0.2f;
  if (unl && pos1) ps = 0.8f;
  if (unl && neg1) ps = 0.2f;
  bool recheck = unl && !pos1 && !neg1;
  float p2 = P2[i];
  bool pos2 = p2 > 0.8f, neg2 = p2 < 0.2f;
  if (recheck && pos2) ps = 0.8f;
  if (recheck && neg2) ps = 0.2f;
  bool confuse = recheck && !pos2 && !neg2;
  if (confuse && p2 > 0.5f) ps = -1.0f;
  if (confuse && p2 < 0.5f) ps = -2.0f;
  out[i] = ps;
}

// ---------------------------------------------------------------------------
// Host side
// ---------------------------------------------------------------------------
static inline dim3 gemm_grid(int M, int N) {
  return dim3((N + BN - 1) / BN, (M + BM - 1) / BM);
}

extern "C" void kernel_launch(void* const* d_in, const int* in_sizes, int n_in,
                              void* d_out, int out_size, void* d_ws, size_t ws_size,
                              hipStream_t stream) {
  (void)in_sizes; (void)n_in; (void)out_size; (void)ws_size;

  // inputs 0..7 in setup_inputs dict order
  const float* p_x        = (const float*)d_in[0];   // HW x 2112
  const float* sp_labels  = (const float*)d_in[1];   // 200 x 2
  const int*   sp_maps    = (const int*)d_in[2];     // HW
  const int*   base_idx   = (const int*)d_in[3];     // 100
  const int*   rel_sel    = (const int*)d_in[4];     // 400
  const int*   rel_sec    = (const int*)d_in[5];     // 400
  const int*   pix_idx    = (const int*)d_in[6];     // HW
  const int*   rec_idx    = (const int*)d_in[7];     // HW
  // params flattened in sorted-key (jax pytree) order:
  const float* cls_b = (const float*)d_in[8];
  const float* cls_w = (const float*)d_in[9];
  const float* fc_b1 = (const float*)d_in[10];
  const float* fc_b2 = (const float*)d_in[11];
  const float* fc_b3 = (const float*)d_in[12];
  const float* fc_w1 = (const float*)d_in[13];
  const float* fc_w2 = (const float*)d_in[14];
  const float* fc_w3 = (const float*)d_in[15];
  const float* fr_b1 = (const float*)d_in[16];
  const float* fr_b2 = (const float*)d_in[17];
  const float* fr_b3 = (const float*)d_in[18];
  const float* fr_w1 = (const float*)d_in[19];
  const float* fr_w2 = (const float*)d_in[20];
  const float* fr_w3 = (const float*)d_in[21];
  const float* rc_b1 = (const float*)d_in[22];
  const float* rc_b2 = (const float*)d_in[23];
  const float* rc_w1 = (const float*)d_in[24];
  const float* rc_w2 = (const float*)d_in[25];

  float* out = (float*)d_out;  // [pred HW][pseudo HW][rel_pred 400][rel_target 400]

  // workspace layout (floats); total ~521 MB
  float* A0    = (float*)d_ws;                        // HW x 1024 (ping)
  float* A1    = A0  + (size_t)HW_ * INT_;            // HW x 1024 (pong)
  float* RF    = A1  + (size_t)HW_ * INT_;            // HW x 512  rel_feat
  float* H3    = RF  + (size_t)HW_ * DREL_;           // HW x 32
  float* SP    = H3  + (size_t)HW_ * DCLS_;           // 1000 x 512
  float* CNT   = SP  + (size_t)NSP_ * DREL_;          // 1000
  float* BF    = CNT + NSP_;                          // 100 x 512
  float* CC400 = BF  + (size_t)NB_ * DREL_;           // 400 x 1024
  float* H400  = CC400 + (size_t)TRN_ * 2 * DREL_;    // 400 x 512
  float* P1    = H400  + (size_t)TRN_ * DREL_;        // HW
  float* P2    = P1 + HW_;                            // HW

  const dim3 blk(256);

  // --- classifier branch ---
  gemm_bias_relu<<<gemm_grid(HW_, INT_),  blk, 0, stream>>>(p_x, fc_w1, fc_b1, A0, HW_, INT_,  CSUM_);
  gemm_bias_relu<<<gemm_grid(HW_, INT_),  blk, 0, stream>>>(A0,  fc_w2, fc_b2, A1, HW_, INT_,  INT_);
  gemm_bias_relu<<<gemm_grid(HW_, DCLS_), blk, 0, stream>>>(A1,  fc_w3, fc_b3, H3, HW_, DCLS_, INT_);
  head2_softmax<<<(HW_ + 255) / 256, blk, 0, stream>>>(H3, cls_w, cls_b, out, HW_, DCLS_);

  // --- relation-feature branch ---
  gemm_bias_relu<<<gemm_grid(HW_, INT_),  blk, 0, stream>>>(p_x, fr_w1, fr_b1, A0, HW_, INT_,  CSUM_);
  gemm_bias_relu<<<gemm_grid(HW_, INT_),  blk, 0, stream>>>(A0,  fr_w2, fr_b2, A1, HW_, INT_,  INT_);
  gemm_bias_relu<<<gemm_grid(HW_, DREL_), blk, 0, stream>>>(A1,  fr_w3, fr_b3, RF, HW_, DREL_, INT_);

  // --- superpixel mean features ---
  zero_f32<<<(NSP_ * DREL_ + NSP_ + 255) / 256, blk, 0, stream>>>(SP, NSP_ * DREL_ + NSP_);
  seg_accum<<<HW_ / 8, blk, 0, stream>>>(RF, sp_maps, SP, CNT);
  seg_div<<<(NSP_ * DREL_ + 255) / 256, blk, 0, stream>>>(SP, CNT, NSP_ * DREL_);
  gather_base<<<(NB_ * DREL_ + 255) / 256, blk, 0, stream>>>(SP, base_idx, BF);

  // --- relation_pred over 400 training pairs ---
  concat_rel<<<(TRN_ * 2 * DREL_ + 255) / 256, blk, 0, stream>>>(BF, SP, rel_sel, rel_sec, CC400);
  gemm_bias_relu<<<gemm_grid(TRN_, DREL_), blk, 0, stream>>>(CC400, rc_w1, rc_b1, H400, TRN_, DREL_, 2 * DREL_);
  head2_softmax<<<(TRN_ + 255) / 256, blk, 0, stream>>>(H400, rc_w2, rc_b2, out + 2 * HW_, TRN_, DREL_);
  rel_target<<<(TRN_ + 255) / 256, blk, 0, stream>>>(sp_labels, rel_sec, out + 2 * HW_ + TRN_);

  // --- pred1 over all pixels ---
  const long long ncc4 = (long long)HW_ * 256;  // float4 elements
  concat_pix<<<(unsigned)((ncc4 + 255) / 256), blk, 0, stream>>>(
      (const float4*)BF, (const float4*)RF, pix_idx, (float4*)A0);
  gemm_bias_relu<<<gemm_grid(HW_, DREL_), blk, 0, stream>>>(A0, rc_w1, rc_b1, A1, HW_, DREL_, 2 * DREL_);
  head2_softmax<<<(HW_ + 255) / 256, blk, 0, stream>>>(A1, rc_w2, rc_b2, P1, HW_, DREL_);

  // --- pred2 (recheck) over all pixels ---
  concat_pix<<<(unsigned)((ncc4 + 255) / 256), blk, 0, stream>>>(
      (const float4*)BF, (const float4*)RF, rec_idx, (float4*)A0);
  gemm_bias_relu<<<gemm_grid(HW_, DREL_), blk, 0, stream>>>(A0, rc_w1, rc_b1, A1, HW_, DREL_, 2 * DREL_);
  head2_softmax<<<(HW_ + 255) / 256, blk, 0, stream>>>(A1, rc_w2, rc_b2, P2, HW_, DREL_);

  // --- pseudo labels ---
  pseudo_labels<<<(HW_ + 255) / 256, blk, 0, stream>>>(sp_maps, sp_labels, P1, P2, out + HW_);
}
